// Unet_2516850835583
// MI455X (gfx1250) — matrix operand order repaired and checked by reading, not compiled, not run on hardware
//
#include <hip/hip_runtime.h>

// ---------------------------------------------------------------------------
// CDNA5 (gfx1250) spherical U-Net forward.
// GEMMs run on v_wmma_f32_16x16x32_bf16 (wave32), gather fused into A-staging.
// Block tile: 32 rows x 64 cols, 4 waves; each wave does 2 WMMAs per K-chunk
// reusing one B fragment. LDS double-buffered: stage chunk c+1 under the
// WMMAs of chunk c; one barrier per chunk.
// ---------------------------------------------------------------------------

typedef __attribute__((ext_vector_type(16))) __bf16 v16bf;
typedef __attribute__((ext_vector_type(8)))  __bf16 v8bf;
typedef __attribute__((ext_vector_type(8)))  float  v8f;

// ---------------------------------------------------------------------------
// Fused gather + GEMM:  Y[M,Cout] = gather(X, neigh)[M, K=7*Cin] @ Wb^T + bias
// GATHER=false: plain GEMM, A row r = X[r*K ...] (up-conv path; K % 32 == 0).
// Wb is bf16, row-major [Cout, Kpad], K-padded with zeros (Kpad % 32 == 0).
// Cin is a power of two -> cinShift = log2(Cin), gather index = shift/mask.
// ---------------------------------------------------------------------------
template <bool GATHER>
__global__ __launch_bounds__(128) void k_conv_gemm_bf16(
    const float* __restrict__ X, const int* __restrict__ neigh,
    const __bf16* __restrict__ Wb, const float* __restrict__ bias,
    float* __restrict__ Y, int M, int K, int Kpad, int cinShift, int Cout)
{
    __shared__ __bf16 As[2][32 * 32];

    const int lane = threadIdx.x & 31;
    const int wave = threadIdx.x >> 5;
    const int hi   = lane >> 4;        // half-wave selector (K/M halves)
    const int l15  = lane & 15;
    const int m0   = blockIdx.x * 32;
    const int n0   = blockIdx.y * 64 + wave * 16;
    const int n    = n0 + l15;         // output column this lane owns
    const int cinMask = (1 << cinShift) - 1;

    // This thread stages 8 consecutive k of one row (4 threads per row).
    const int srow   = m0 + (int)(threadIdx.x >> 2);       // staged row
    const int scol   = (int)(threadIdx.x & 3) << 3;        // first staged k
    const bool rowok = srow < M;
    const int nbase  = srow * 7;

    // Stage one 32x32 A chunk (f32 -> bf16) into buffer `buf`.
    auto stage = [&](int kb, int buf) {
        float vals[8];
        if (rowok) {
            if (GATHER) {
                int k0 = kb + scol;
                if (cinShift >= 3) {
                    // 8 consecutive k touch at most 2 ring neighbors.
                    int j0 = k0 >> cinShift;       if (j0 > 6) j0 = 6;
                    int j1 = (k0 + 7) >> cinShift; if (j1 > 6) j1 = 6;
                    int nb0 = neigh[nbase + j0];
                    int nb1 = neigh[nbase + j1];
                    #pragma unroll
                    for (int t = 0; t < 8; ++t) {
                        int k  = k0 + t;
                        int j  = k >> cinShift;
                        int nb = (j == j0) ? nb0 : nb1;
                        vals[t] = X[((size_t)nb << cinShift) + (k & cinMask)];
                    }
                } else {
                    #pragma unroll
                    for (int t = 0; t < 8; ++t) {
                        int k = k0 + t;
                        int j = k >> cinShift; if (j > 6) j = 6;
                        vals[t] = X[((size_t)neigh[nbase + j] << cinShift) +
                                    (k & cinMask)];
                    }
                }
                #pragma unroll
                for (int t = 0; t < 8; ++t)
                    if (k0 + t >= K) vals[t] = 0.0f;
            } else {
                // K % 32 == 0 here: no tail, 32B-aligned float4 pairs.
                const float4* p =
                    (const float4*)(X + (size_t)srow * K + kb + scol);
                float4 u = p[0];
                float4 w = p[1];
                vals[0] = u.x; vals[1] = u.y; vals[2] = u.z; vals[3] = u.w;
                vals[4] = w.x; vals[5] = w.y; vals[6] = w.z; vals[7] = w.w;
            }
        } else {
            #pragma unroll
            for (int t = 0; t < 8; ++t) vals[t] = 0.0f;
        }
        v8bf pk;
        #pragma unroll
        for (int t = 0; t < 8; ++t) pk[t] = (__bf16)vals[t];
        *(v8bf*)(&As[buf][threadIdx.x << 3]) = pk;   // one ds_store_b128
    };

    v8f acc0 = {};
    v8f acc1 = {};

    stage(0, 0);
    int it = 0;
    for (int kb = 0; kb < Kpad; kb += 32, ++it) {
        __syncthreads();                 // staging of current buffer complete
        const int cb = it & 1;
        if (kb + 32 < Kpad)
            stage(kb + 32, cb ^ 1);      // overlap next stage with WMMAs

        // ---- B fragment, contiguous k order (same as A) ---------------------
        // lanes 0-15: K kb+0..15 ; lanes 16-31: K kb+16..31 ; col = l15.
        v16bf b = {};
        if (n < Cout) {
            const __bf16* wp = Wb + (size_t)n * Kpad + kb + hi * 16;
            b = *(const v16bf*)wp;              // 32B aligned (Kpad%32==0)
            __builtin_prefetch(wp + 32, 0, 1);  // next K-chunk
        }

        // ---- A fragments, contiguous k order (same as B) --------------------
        // lanes 0-15 : elems 0..15 -> K kb+0..15
        // lanes16-31 : elems 0..15 -> K kb+16..31
        v16bf a0, a1;
        #pragma unroll
        for (int e = 0; e < 16; ++e) {
            int koff = e + hi * 16;
            a0[e] = As[cb][l15 * 32 + koff];
            a1[e] = As[cb][(16 + l15) * 32 + koff];
        }

        acc0 = __builtin_amdgcn_wmma_f32_16x16x32_bf16(
                   false, a0, false, b, (short)0, acc0, false, false);
        acc1 = __builtin_amdgcn_wmma_f32_16x16x32_bf16(
                   false, a1, false, b, (short)0, acc1, false, false);
    }

    // ---- store D (VGPR r: rows +r for lanes 0-15, +8+r for lanes 16-31) ---
    float bv = (n < Cout) ? bias[n] : 0.0f;
    #pragma unroll
    for (int r = 0; r < 8; ++r) {
        int row0 = m0 + r + hi * 8;
        int row1 = row0 + 16;
        if (row0 < M && n < Cout)
            Y[(size_t)row0 * Cout + n] = acc0[r] + bv;
        if (row1 < M && n < Cout)
            Y[(size_t)row1 * Cout + n] = acc1[r] + bv;
    }
}

// ---------------------------------------------------------------------------
// BatchNorm training-mode statistics: one block per channel column.
// ---------------------------------------------------------------------------
__global__ __launch_bounds__(256) void k_bn_stats(
    const float* __restrict__ X, int Nr, int C,
    float* __restrict__ mean, float* __restrict__ rstd)
{
    __shared__ float s1[256];
    __shared__ float s2[256];
    int c = blockIdx.x;
    float a = 0.0f, b = 0.0f;
    for (int r = threadIdx.x; r < Nr; r += 256) {
        float v = X[(size_t)r * C + c];
        a += v;
        b += v * v;
    }
    s1[threadIdx.x] = a;
    s2[threadIdx.x] = b;
    __syncthreads();
    for (int s = 128; s > 0; s >>= 1) {
        if ((int)threadIdx.x < s) {
            s1[threadIdx.x] += s1[threadIdx.x + s];
            s2[threadIdx.x] += s2[threadIdx.x + s];
        }
        __syncthreads();
    }
    if (threadIdx.x == 0) {
        float m = s1[0] / (float)Nr;
        float v = s2[0] / (float)Nr - m * m;   // biased variance
        mean[c] = m;
        rstd[c] = rsqrtf(v + 1e-5f);
    }
}

// ---------------------------------------------------------------------------
// In-place BN affine + LeakyReLU(0.2).  C is a power of two (cShift=log2 C).
// ---------------------------------------------------------------------------
__global__ __launch_bounds__(256) void k_bn_lrelu(
    float* __restrict__ X, int Nr, int cShift,
    const float* __restrict__ mean, const float* __restrict__ rstd,
    const float* __restrict__ g, const float* __restrict__ be)
{
    size_t i = (size_t)blockIdx.x * 256 + threadIdx.x;
    if (i >= ((size_t)Nr << cShift)) return;
    int c = (int)(i & (((size_t)1 << cShift) - 1));
    float y = g[c] * (X[i] - mean[c]) * rstd[c] + be[c];
    X[i] = (y > 0.0f) ? y : 0.2f * y;
}

// ---------------------------------------------------------------------------
// 7-neighbor mean pool:  Y[r,c] = mean_j X[neigh[r*7+j], c];  C = 1<<cShift.
// ---------------------------------------------------------------------------
__global__ __launch_bounds__(256) void k_pool7(
    const float* __restrict__ X, const int* __restrict__ ng,
    float* __restrict__ Y, int Nc, int cShift)
{
    size_t i = (size_t)blockIdx.x * 256 + threadIdx.x;
    if (i >= ((size_t)Nc << cShift)) return;
    int r = (int)(i >> cShift);
    int c = (int)(i & ((1 << cShift) - 1));
    float s = 0.0f;
    #pragma unroll
    for (int j = 0; j < 7; ++j)
        s += X[((size_t)ng[r * 7 + j] << cShift) + c];
    Y[i] = s * (1.0f / 7.0f);
}

// ---------------------------------------------------------------------------
// Up-conv combine + skip concat.  C = 1<<cShift.
//   Y7  : [raw*7, C] up-conv output rows
//   Out : [nnew, 2C]; cols [0,C) = x1/x2 recombination, [C,2C) = skip.
// ---------------------------------------------------------------------------
__global__ __launch_bounds__(256) void k_up_combine(
    const float* __restrict__ Y7, const int* __restrict__ top,
    const int* __restrict__ dn, const float* __restrict__ skip,
    float* __restrict__ Out, int raw, int nnew, int cShift)
{
    size_t i = (size_t)blockIdx.x * 256 + threadIdx.x;
    if (i >= ((size_t)nnew << cShift)) return;
    int r = (int)(i >> cShift);
    int c = (int)(i & ((1 << cShift) - 1));
    int C = 1 << cShift;
    float v;
    if (r < raw) {
        v = Y7[((size_t)top[r] << cShift) + c];
    } else {
        int t = r - raw;
        v = 0.5f * (Y7[((size_t)dn[2 * t] << cShift) + c] +
                    Y7[((size_t)dn[2 * t + 1] << cShift) + c]);
    }
    Out[((size_t)r << (cShift + 1)) + c]     = v;
    Out[((size_t)r << (cShift + 1)) + C + c] = skip[i];
}

// ---------------------------------------------------------------------------
// Final Linear 32 -> 2 (too skinny for WMMA).
// ---------------------------------------------------------------------------
__global__ __launch_bounds__(256) void k_final_linear(
    const float* __restrict__ X, const float* __restrict__ W,
    const float* __restrict__ B, float* __restrict__ Out, int Nr)
{
    int r = blockIdx.x * 256 + threadIdx.x;
    if (r >= Nr) return;
    float s0 = B[0], s1 = B[1];
    #pragma unroll
    for (int k = 0; k < 32; ++k) {
        float x = X[(size_t)r * 32 + k];
        s0 += x * W[k];
        s1 += x * W[32 + k];
    }
    Out[(size_t)r * 2 + 0] = s0;
    Out[(size_t)r * 2 + 1] = s1;
}

// ---------------------------------------------------------------------------
// Weight convert f32 -> bf16 with K padded to a multiple of 32 (zeros).
// ---------------------------------------------------------------------------
__global__ __launch_bounds__(256) void k_cvt_pad(
    const float* __restrict__ W, __bf16* __restrict__ Wb,
    int Cout, int K, int Kpad)
{
    size_t i = (size_t)blockIdx.x * 256 + threadIdx.x;
    if (i >= (size_t)Cout * Kpad) return;
    int nrow = (int)(i / Kpad);
    int k    = (int)(i % Kpad);
    Wb[i] = (__bf16)((k < K) ? W[(size_t)nrow * K + k] : 0.0f);
}

// ===========================================================================
// Host orchestration
// ===========================================================================
static inline int ilog2(int v) { int s = 0; while ((1 << s) < v) ++s; return s; }

extern "C" void kernel_launch(void* const* d_in, const int* in_sizes, int n_in,
                              void* d_out, int out_size, void* d_ws, size_t ws_size,
                              hipStream_t stream)
{
    (void)in_sizes; (void)n_in; (void)out_size; (void)ws_size;

    static const int NSv[5] = {163842, 40962, 10242, 2562, 642};
    static const int CHv[6] = {2, 32, 64, 128, 256, 512};

    // ---- input layout: JAX pytree flatten (dict keys sorted) ---------------
    // top-level sorted: idx, params, x
    //  idx: down[0..3](0-3), neigh[0..4](4-8), top[0..3](9-12)
    //  params: down[i]{c1{b,be,g,w},c2{b,be,g,w}} (13..52),
    //          out{b(53), w(54)},
    //          up[j]{c1{b,be,g,w},c2{...},up{b,w}} (55..94)
    //  x: 95
    const int* IDX_DOWN[4];
    const int* IDX_NEIGH[5];
    const int* IDX_TOP[4];
    for (int j = 0; j < 4; ++j) IDX_DOWN[j]  = (const int*)d_in[0 + j];
    for (int i = 0; i < 5; ++i) IDX_NEIGH[i] = (const int*)d_in[4 + i];
    for (int j = 0; j < 4; ++j) IDX_TOP[j]   = (const int*)d_in[9 + j];

    struct CBNRaw { const float *b, *be, *g, *w; };
    CBNRaw dn_raw[5][2];
    for (int i = 0; i < 5; ++i) {
        int base = 13 + i * 8;
        dn_raw[i][0] = { (const float*)d_in[base + 0], (const float*)d_in[base + 1],
                         (const float*)d_in[base + 2], (const float*)d_in[base + 3] };
        dn_raw[i][1] = { (const float*)d_in[base + 4], (const float*)d_in[base + 5],
                         (const float*)d_in[base + 6], (const float*)d_in[base + 7] };
    }
    const float* out_b = (const float*)d_in[53];
    const float* out_w = (const float*)d_in[54];
    CBNRaw up_raw[4][2];
    const float *upl_b[4], *upl_w[4];
    for (int j = 0; j < 4; ++j) {
        int base = 55 + j * 10;
        up_raw[j][0] = { (const float*)d_in[base + 0], (const float*)d_in[base + 1],
                         (const float*)d_in[base + 2], (const float*)d_in[base + 3] };
        up_raw[j][1] = { (const float*)d_in[base + 4], (const float*)d_in[base + 5],
                         (const float*)d_in[base + 6], (const float*)d_in[base + 7] };
        upl_b[j] = (const float*)d_in[base + 8];
        upl_w[j] = (const float*)d_in[base + 9];
    }
    const float* x_in = (const float*)d_in[95];

    // ---- workspace bump allocator ------------------------------------------
    char* ws = (char*)d_ws;
    size_t off = 0;
    auto alloc = [&](size_t bytes) -> void* {
        void* p = ws + off;
        off = (off + bytes + 255) & ~(size_t)255;
        return p;
    };

    float* d_mean = (float*)alloc(512 * sizeof(float));
    float* d_rstd = (float*)alloc(512 * sizeof(float));
    float* D[4];                                   // skip buffers (down outputs)
    D[0] = (float*)alloc((size_t)NSv[0] * 32  * sizeof(float));
    D[1] = (float*)alloc((size_t)NSv[1] * 64  * sizeof(float));
    D[2] = (float*)alloc((size_t)NSv[2] * 128 * sizeof(float));
    D[3] = (float*)alloc((size_t)NSv[3] * 256 * sizeof(float));
    float* bufA = (float*)alloc((size_t)NSv[0] * 64 * sizeof(float));
    float* bufB = (float*)alloc((size_t)NSv[0] * 64 * sizeof(float));
    float* bufC = (float*)alloc((size_t)NSv[1] * 7 * 32 * sizeof(float)); // y / pool

    // ---- weight prep: convert to bf16, pad K to multiple of 32 -------------
    auto prep_w = [&](const float* w, int cout, int K, int* KpadOut) -> __bf16* {
        int Kpad = (K + 31) & ~31;
        *KpadOut = Kpad;
        __bf16* wb = (__bf16*)alloc((size_t)cout * Kpad * sizeof(__bf16));
        size_t tot = (size_t)cout * Kpad;
        k_cvt_pad<<<(unsigned)((tot + 255) / 256), 256, 0, stream>>>(w, wb, cout, K, Kpad);
        return wb;
    };

    struct Conv { const float *b, *be, *g; __bf16* wb; int cin, cout, K, Kpad; };
    Conv DNc[5][2];
    for (int i = 0; i < 5; ++i) {
        int cout = CHv[i + 1];
        for (int s = 0; s < 2; ++s) {
            int cin = (s == 0) ? ((i == 0) ? 2 : CHv[i]) : cout;
            Conv L;
            L.b = dn_raw[i][s].b; L.be = dn_raw[i][s].be; L.g = dn_raw[i][s].g;
            L.cin = cin; L.cout = cout; L.K = 7 * cin;
            L.wb = prep_w(dn_raw[i][s].w, cout, L.K, &L.Kpad);
            DNc[i][s] = L;
        }
    }
    Conv UPc[4][2];
    for (int j = 0; j < 4; ++j) {
        int Clo = CHv[4 - j];
        for (int s = 0; s < 2; ++s) {
            int cin = (s == 0) ? 2 * Clo : Clo;
            Conv L;
            L.b = up_raw[j][s].b; L.be = up_raw[j][s].be; L.g = up_raw[j][s].g;
            L.cin = cin; L.cout = Clo; L.K = 7 * cin;
            L.wb = prep_w(up_raw[j][s].w, Clo, L.K, &L.Kpad);
            UPc[j][s] = L;
        }
    }
    struct Lin { const float* b; __bf16* wb; int K, Kpad, cout; };
    Lin UPl[4];
    for (int j = 0; j < 4; ++j) {
        UPl[j].b = upl_b[j];
        UPl[j].K = CHv[5 - j];             // input channels (C_hi), % 32 == 0
        UPl[j].cout = 7 * CHv[4 - j];      // output features (7*C_lo)
        UPl[j].wb = prep_w(upl_w[j], UPl[j].cout, UPl[j].K, &UPl[j].Kpad);
    }

    // ---- launch helpers ----------------------------------------------------
    auto gemm_gather = [&](const float* X, const int* ng, const Conv& L,
                           float* Y, int M) {
        dim3 grid((unsigned)((M + 31) / 32), (unsigned)((L.cout + 63) / 64));
        k_conv_gemm_bf16<true><<<grid, 128, 0, stream>>>(
            X, ng, L.wb, L.b, Y, M, L.K, L.Kpad, ilog2(L.cin), L.cout);
    };
    auto gemm_plain = [&](const float* X, const Lin& L, float* Y, int M) {
        dim3 grid((unsigned)((M + 31) / 32), (unsigned)((L.cout + 63) / 64));
        k_conv_gemm_bf16<false><<<grid, 128, 0, stream>>>(
            X, nullptr, L.wb, L.b, Y, M, L.K, L.Kpad, 0, L.cout);
    };
    auto conv_bn = [&](const float* X, const int* ng, const Conv& L,
                       float* Y, int M) {
        gemm_gather(X, ng, L, Y, M);
        k_bn_stats<<<(unsigned)L.cout, 256, 0, stream>>>(Y, M, L.cout, d_mean, d_rstd);
        size_t tot = (size_t)M * L.cout;
        k_bn_lrelu<<<(unsigned)((tot + 255) / 256), 256, 0, stream>>>(
            Y, M, ilog2(L.cout), d_mean, d_rstd, L.g, L.be);
    };

    // ---- down path ---------------------------------------------------------
    conv_bn(x_in, IDX_NEIGH[0], DNc[0][0], bufA, NSv[0]);
    conv_bn(bufA, IDX_NEIGH[0], DNc[0][1], D[0], NSv[0]);
    float* cur = D[0];
    for (int i = 1; i < 5; ++i) {
        int M  = NSv[i];
        int Cp = CHv[i];                   // channels being pooled
        size_t tot = (size_t)M * Cp;
        k_pool7<<<(unsigned)((tot + 255) / 256), 256, 0, stream>>>(
            cur, IDX_NEIGH[i - 1], bufC, M, ilog2(Cp));
        conv_bn(bufC, IDX_NEIGH[i], DNc[i][0], bufA, M);
        float* dst = (i < 4) ? D[i] : bufB;
        conv_bn(bufA, IDX_NEIGH[i], DNc[i][1], dst, M);
        cur = dst;
    }

    // ---- up path -----------------------------------------------------------
    for (int j = 0; j < 4; ++j) {
        int raw  = NSv[4 - j];
        int nnew = NSv[3 - j];
        int Clo  = CHv[4 - j];
        // y = cur @ up.w^T + b  -> bufC viewed as [raw*7, Clo]
        gemm_plain(cur, UPl[j], bufC, raw);
        float* cat = (cur == bufA) ? bufB : bufA;
        size_t tot = (size_t)nnew * Clo;
        k_up_combine<<<(unsigned)((tot + 255) / 256), 256, 0, stream>>>(
            bufC, IDX_TOP[j], IDX_DOWN[j], D[3 - j], cat, raw, nnew, ilog2(Clo));
        float* t = (cat == bufA) ? bufB : bufA;
        conv_bn(cat, IDX_NEIGH[3 - j], UPc[j][0], t, nnew);
        conv_bn(t,   IDX_NEIGH[3 - j], UPc[j][1], cat, nnew);
        cur = cat;
    }

    // ---- final linear ------------------------------------------------------
    k_final_linear<<<(unsigned)((NSv[0] + 255) / 256), 256, 0, stream>>>(
        cur, out_w, out_b, (float*)d_out, NSv[0]);
}
